// VectorAttention_59579786330695
// MI455X (gfx1250) — matrix operand order repaired and checked by reading, not compiled, not run on hardware
//
#include <hip/hip_runtime.h>

#define B_    8
#define T_    256
#define TOKD  128
#define TPRD  9
#define INNER 64
#define HID   32
#define QKVW  (3*INNER)   // 192

typedef __attribute__((ext_vector_type(16))) _Float16 v16h;
typedef __attribute__((ext_vector_type(8)))  _Float16 v8h;
typedef __attribute__((ext_vector_type(8)))  float    v8f;
typedef __attribute__((ext_vector_type(4)))  unsigned int u32x4;
typedef __attribute__((ext_vector_type(8)))  unsigned int u32x8;

union FragA { v16h v; v8h h[2]; };
union H8   { v8h v; _Float16 e[8]; };

__device__ inline v8f wmma32f16(v16h a, v16h b, v8f c) {
  return __builtin_amdgcn_wmma_f32_16x16x32_f16(false, a, false, b, (short)0, c, false, false);
}

// ---------------- stage 1a: qkv = x @ w_qkv (f32) ----------------
__global__ void qkv_kernel(const float* __restrict__ x, const float* __restrict__ w,
                           float* __restrict__ qkv) {
  int idx = blockIdx.x * 256 + threadIdx.x;
  if (idx >= B_ * T_ * QKVW) return;
  int col = idx % QKVW;
  int row = idx / QKVW;
  const float* xr = x + (size_t)row * TOKD;
  float acc = 0.f;
  #pragma unroll 8
  for (int d = 0; d < TOKD; ++d) acc = fmaf(xr[d], w[d * QKVW + col], acc);
  qkv[idx] = acc;
}

// -------- stage 1b: pack weight B-fragments (f16) + fold BN --------
// B-fragment layout for v_wmma_f32_16x16x32_f16:
//   lane L holds column n = nt*16 + (L&15), two K-runs of 8 at
//   kbase = kt*32 + (L>>4)*8, e<8 -> K = kbase+e, e>=8 -> K = kbase+8+e.
__global__ void pack_kernel(const float* __restrict__ w1, const float* __restrict__ w2,
                            const float* __restrict__ aw1, const float* __restrict__ aw2,
                            const float* __restrict__ b1t, const float* __restrict__ gamma,
                            const float* __restrict__ beta, const float* __restrict__ mean,
                            const float* __restrict__ var,
                            _Float16* __restrict__ fW1, _Float16* __restrict__ fW2,
                            _Float16* __restrict__ fA1, _Float16* __restrict__ fA2,
                            float* __restrict__ scaleC, float* __restrict__ shiftC) {
  int t = threadIdx.x;
  if (t < 704) {
    int tg = t / 32, lane = t % 32;
    const float* src; _Float16* dst; int nt, kt, Kdim, ldN;
    if (tg < 2)       { src = w1;  dst = fW1 + tg * 512;  nt = tg;  kt = 0;  Kdim = TPRD; ldN = HID; }
    else if (tg < 6)  { int q = tg - 2;  src = w2;  dst = fW2 + q * 512; nt = q;     kt = 0;     Kdim = HID;   ldN = INNER; }
    else if (tg < 14) { int q = tg - 6;  src = aw1; dst = fA1 + q * 512; nt = q / 2; kt = q % 2; Kdim = INNER; ldN = INNER; }
    else              { int q = tg - 14; src = aw2; dst = fA2 + q * 512; nt = q / 2; kt = q % 2; Kdim = INNER; ldN = INNER; }
    int n = nt * 16 + (lane & 15);
    int kbase = kt * 32 + ((lane >> 4) << 3);
    #pragma unroll
    for (int e = 0; e < 16; ++e) {
      int K = kbase + e + ((e >> 3) << 3);
      float val = (K < Kdim) ? src[K * ldN + n] : 0.f;
      dst[lane * 16 + e] = (_Float16)val;
    }
  } else if (t < 736) {
    int hc = t - 704;
    float s  = rsqrtf(var[hc] + 1e-5f);
    float sg = gamma[hc] * s;
    scaleC[hc] = sg;
    shiftC[hc] = beta[hc] + (b1t[hc] - mean[hc]) * sg;
  }
}

// -------- stage 2: fused per-(b,i) attention, 8 waves, all-WMMA pipeline --------
__global__ __launch_bounds__(256) void fused_kernel(
    const float* __restrict__ r, const float* __restrict__ qkv,
    const float* __restrict__ b2t,
    const float* __restrict__ ab1, const float* __restrict__ ab2,
    const _Float16* __restrict__ fW1, const _Float16* __restrict__ fW2,
    const _Float16* __restrict__ fA1, const _Float16* __restrict__ fA2,
    const float* __restrict__ scaleC, const float* __restrict__ shiftC,
    float* __restrict__ out) {
  __shared__ float    sE[T_ * INNER];   // rel_emb, then vj, f32 (64 KB)
  __shared__ _Float16 sR16[T_ * HID];   // r, f16, K-padded      (16 KB)
  __shared__ _Float16 sH[T_ * HID];     // BN/ReLU(h), f16       (16 KB)
  __shared__ _Float16 sT[T_ * INNER];   // r-stage, T, U in place(32 KB)
  __shared__ float    sScale[HID], sShift[HID];
  __shared__ float    sRedA[8 * 64], sRedB[8 * 64], sColMax[64];

  const int tid  = threadIdx.x;
  const int lane = tid & 31;
  const int wv   = tid >> 5;
  const int half = lane >> 4;
  const int l16  = lane & 15;
  const int bi   = blockIdx.x;          // b*T + i
  const int b    = bi / T_;

  if (tid < HID) { sScale[tid] = scaleC[tid]; sShift[tid] = shiftC[tid]; }

  // ---- phase -1: TDM async copy of r[b,i,:,:] (9216 B contiguous) into LDS ----
  // D# per CDNA5 ISA: 1-D tile, data_size=4B, tile_dim0=tensor_dim0=2304.
  float* sR32 = (float*)sT;             // staging overlay; sT unused until phase 3
  if (wv == 0) {
    unsigned long long gaddr = (unsigned long long)(const void*)(r + (size_t)bi * T_ * TPRD);
    unsigned int       laddr = (unsigned int)(unsigned long long)(void*)sR32;
    u32x4 g0 = { 1u,                                    // count=1 valid descriptor
                 laddr,                                 // lds_addr[31:0]
                 (unsigned int)(gaddr & 0xffffffffu),   // global_addr[31:0]
                 (unsigned int)((gaddr >> 32) & 0x01ffffffu) | 0x80000000u }; // addr[56:32] | type=2
    u32x8 g1 = { 0x00020000u,   // workgroup_mask=0, data_size=2 (4B)
                 0x09000000u,   // tensor_dim0[15:0]=2304 @ bits 63:48
                 0x00010000u,   // tensor_dim0[31:16]=0, tensor_dim1[15:0]=1
                 0x09000000u,   // tensor_dim1[31:16]=0, tile_dim0=2304 @ bits 127:112
                 0x00000001u,   // tile_dim1=1, tile_dim2=0
                 2304u,         // tensor_dim0_stride[31:0]
                 0u, 0u };
    asm volatile("tensor_load_to_lds %0, %1" :: "s"(g0), "s"(g1) : "memory");
    asm volatile("s_wait_tensorcnt 0x0" ::: "memory");
  }
  __syncthreads();

  // ---- phase 0: r row -> f16, zero-padded K=9 -> 32 (thread = row j) ----
  {
    const int j = tid;
    const float* rr = sR32 + j * TPRD;
    v8h z = {};
    H8 c0, c1;
    #pragma unroll
    for (int d = 0; d < 8; ++d) c0.e[d] = (_Float16)rr[d];
    c1.v = z;
    c1.e[0] = (_Float16)rr[8];
    *(v8h*)(sR16 + j * HID)      = c0.v;
    *(v8h*)(sR16 + j * HID + 8)  = c1.v;
    *(v8h*)(sR16 + j * HID + 16) = z;
    *(v8h*)(sR16 + j * HID + 24) = z;
  }
  __syncthreads();

  // ---- phase 1: H = relu(scale*(r@w1) + shift) via WMMA (K=32 padded) ----
  #pragma unroll
  for (int rtl = 0; rtl < 2; ++rtl) {
    const int rt  = 2 * wv + rtl;
    const int row = rt * 16 + l16;
    FragA a;
    a.h[0] = *(const v8h*)(sR16 + row * HID + half * 8);
    a.h[1] = *(const v8h*)(sR16 + row * HID + 16 + half * 8);
    #pragma unroll
    for (int nt = 0; nt < 2; ++nt) {
      v16h bf = *(const v16h*)(fW1 + nt * 512 + lane * 16);
      v8f acc = {};
      acc = wmma32f16(a.v, bf, acc);
      const int n  = nt * 16 + l16;
      const float sc = sScale[n], sh = sShift[n];
      #pragma unroll
      for (int rr_ = 0; rr_ < 8; ++rr_) {
        int m = rt * 16 + rr_ + 8 * half;
        sH[m * HID + n] = (_Float16)fmaxf(fmaf(acc[rr_], sc, sh), 0.f);
      }
    }
  }
  __syncthreads();

  // ---- phase 2: E = H @ W2 + b2t  (WMMA, K=32) ----
  #pragma unroll
  for (int rtl = 0; rtl < 2; ++rtl) {
    const int rt  = 2 * wv + rtl;
    const int row = rt * 16 + l16;
    FragA a;
    a.h[0] = *(const v8h*)(sH + row * HID + half * 8);
    a.h[1] = *(const v8h*)(sH + row * HID + 16 + half * 8);
    #pragma unroll
    for (int nt = 0; nt < 4; ++nt) {
      v16h bf = *(const v16h*)(fW2 + nt * 512 + lane * 16);
      v8f acc = {};
      acc = wmma32f16(a.v, bf, acc);
      const int n = nt * 16 + l16;
      const float bias = b2t[n];
      #pragma unroll
      for (int rr_ = 0; rr_ < 8; ++rr_) {
        int m = rt * 16 + rr_ + 8 * half;
        sE[m * INNER + n] = acc[rr_] + bias;
      }
    }
  }
  __syncthreads();

  // ---- phase 3: sT = f16(q_i - k_j + E);  sE <- vj = v_j + E  (coalesced) ----
  {
    const int c  = tid & 63;            // channel, constant per thread
    const int j0 = tid >> 6;            // 4 rows per iteration
    const float qc = qkv[(size_t)bi * QKVW + c];
    #pragma unroll 4
    for (int it = 0; it < 64; ++it) {
      const int j = it * 4 + j0;
      const float* row = qkv + ((size_t)b * T_ + j) * QKVW;
      float kv = row[INNER + c];
      float vv = row[2 * INNER + c];
      float Ev = sE[j * INNER + c];
      sT[j * INNER + c] = (_Float16)(qc - kv + Ev);
      sE[j * INNER + c] = Ev + vv;
    }
  }
  __syncthreads();

  // ---- phase 4: U = relu(T @ aw1 + ab1), f16, in-place (A frags in regs) ----
  #pragma unroll
  for (int rtl = 0; rtl < 2; ++rtl) {
    const int rt  = 2 * wv + rtl;
    const int row = rt * 16 + l16;
    FragA a0, a1;
    a0.h[0] = *(const v8h*)(sT + row * INNER + half * 8);
    a0.h[1] = *(const v8h*)(sT + row * INNER + 16 + half * 8);
    a1.h[0] = *(const v8h*)(sT + row * INNER + 32 + half * 8);
    a1.h[1] = *(const v8h*)(sT + row * INNER + 48 + half * 8);
    #pragma unroll
    for (int nt = 0; nt < 4; ++nt) {
      v16h b0 = *(const v16h*)(fA1 + (nt * 2 + 0) * 512 + lane * 16);
      v16h b1 = *(const v16h*)(fA1 + (nt * 2 + 1) * 512 + lane * 16);
      v8f acc = {};
      acc = wmma32f16(a0.v, b0, acc);
      acc = wmma32f16(a1.v, b1, acc);
      const int n = nt * 16 + l16;
      const float bias = ab1[n];
      #pragma unroll
      for (int rr_ = 0; rr_ < 8; ++rr_) {
        int m = rt * 16 + rr_ + 8 * half;
        sT[m * INNER + n] = (_Float16)fmaxf(acc[rr_] + bias, 0.f);
      }
    }
  }
  __syncthreads();

  // ---- phase 5: S = U @ aw2 + ab2, kept in accumulator registers ----
  v8f accS[2][4];
  #pragma unroll
  for (int rtl = 0; rtl < 2; ++rtl) {
    const int rt  = 2 * wv + rtl;
    const int row = rt * 16 + l16;
    FragA a0, a1;
    a0.h[0] = *(const v8h*)(sT + row * INNER + half * 8);
    a0.h[1] = *(const v8h*)(sT + row * INNER + 16 + half * 8);
    a1.h[0] = *(const v8h*)(sT + row * INNER + 32 + half * 8);
    a1.h[1] = *(const v8h*)(sT + row * INNER + 48 + half * 8);
    #pragma unroll
    for (int nt = 0; nt < 4; ++nt) {
      v16h b0 = *(const v16h*)(fA2 + (nt * 2 + 0) * 512 + lane * 16);
      v16h b1 = *(const v16h*)(fA2 + (nt * 2 + 1) * 512 + lane * 16);
      v8f acc = {};
      acc = wmma32f16(a0.v, b0, acc);
      acc = wmma32f16(a1.v, b1, acc);
      const float bias = ab2[nt * 16 + l16];
      #pragma unroll
      for (int rr_ = 0; rr_ < 8; ++rr_) acc[rr_] += bias;
      accS[rtl][nt] = acc;
    }
  }

  // ---- phase 6: softmax over j (rows) per channel, weighted sum ----
  // wave wv holds rows [32*wv, 32*wv+32); row m = 32*wv + rtl*16 + rr + 8*half
  #pragma unroll
  for (int nt = 0; nt < 4; ++nt) {
    float mx = -3.402823e38f;
    #pragma unroll
    for (int rtl = 0; rtl < 2; ++rtl)
      #pragma unroll
      for (int rr_ = 0; rr_ < 8; ++rr_) mx = fmaxf(mx, accS[rtl][nt][rr_]);
    mx = fmaxf(mx, __shfl_xor(mx, 16, 32));
    if (lane < 16) sRedA[wv * 64 + nt * 16 + l16] = mx;
  }
  __syncthreads();
  if (tid < 64) {
    float m = sRedA[tid];
    #pragma unroll
    for (int w2_ = 1; w2_ < 8; ++w2_) m = fmaxf(m, sRedA[w2_ * 64 + tid]);
    sColMax[tid] = m;
  }
  __syncthreads();
  #pragma unroll
  for (int nt = 0; nt < 4; ++nt) {
    const int n  = nt * 16 + l16;
    const float gm = sColMax[n];
    float se = 0.f, sv = 0.f;
    #pragma unroll
    for (int rtl = 0; rtl < 2; ++rtl) {
      #pragma unroll
      for (int rr_ = 0; rr_ < 8; ++rr_) {
        int m = 32 * wv + rtl * 16 + rr_ + 8 * half;
        float e = __expf(accS[rtl][nt][rr_] - gm);
        se += e;
        sv  = fmaf(e, sE[m * INNER + n], sv);   // sE holds vj = v + rel_emb
      }
    }
    se += __shfl_xor(se, 16, 32);
    sv += __shfl_xor(sv, 16, 32);
    if (lane < 16) { sRedA[wv * 64 + n] = se; sRedB[wv * 64 + n] = sv; }
  }
  __syncthreads();
  if (tid < 64) {
    float se = 0.f, sv = 0.f;
    #pragma unroll
    for (int w2_ = 0; w2_ < 8; ++w2_) { se += sRedA[w2_ * 64 + tid]; sv += sRedB[w2_ * 64 + tid]; }
    out[(size_t)bi * INNER + tid] = sv / se;
  }
}

extern "C" void kernel_launch(void* const* d_in, const int* in_sizes, int n_in,
                              void* d_out, int out_size, void* d_ws, size_t ws_size,
                              hipStream_t stream) {
  (void)in_sizes; (void)n_in; (void)out_size; (void)ws_size;
  const float* x      = (const float*)d_in[0];
  const float* r      = (const float*)d_in[1];
  const float* w_qkv  = (const float*)d_in[2];
  const float* tpr_w1 = (const float*)d_in[3];
  const float* tpr_b1 = (const float*)d_in[4];
  const float* gamma  = (const float*)d_in[5];
  const float* beta   = (const float*)d_in[6];
  const float* mean   = (const float*)d_in[7];
  const float* var    = (const float*)d_in[8];
  const float* tpr_w2 = (const float*)d_in[9];
  const float* tpr_b2 = (const float*)d_in[10];
  const float* aw1    = (const float*)d_in[11];
  const float* ab1    = (const float*)d_in[12];
  const float* aw2    = (const float*)d_in[13];
  const float* ab2    = (const float*)d_in[14];

  char* ws = (char*)d_ws;
  float*    qkvp = (float*)ws;                        // 2048*192 f32 = 1572864 B
  _Float16* fW1  = (_Float16*)(ws + 1572864);         // 2 tiles * 1024 B
  _Float16* fW2  = (_Float16*)(ws + 1572864 + 2048);  // 4 tiles * 1024 B
  _Float16* fA1  = (_Float16*)(ws + 1572864 + 6144);  // 8 tiles * 1024 B
  _Float16* fA2  = (_Float16*)(ws + 1572864 + 14336); // 8 tiles * 1024 B
  float*    scC  = (float*)(ws + 1572864 + 22528);
  float*    shC  = scC + HID;

  qkv_kernel<<<(B_ * T_ * QKVW + 255) / 256, 256, 0, stream>>>(x, w_qkv, qkvp);
  pack_kernel<<<1, 1024, 0, stream>>>(tpr_w1, tpr_w2, aw1, aw2, tpr_b1, gamma, beta, mean, var,
                                      fW1, fW2, fA1, fA2, scC, shC);
  fused_kernel<<<B_ * T_, 256, 0, stream>>>(r, qkvp, tpr_b2, ab1, ab2,
                                            fW1, fW2, fA1, fA2, scC, shC, (float*)d_out);
}
